// MolTriangleAttention_50044958933073
// MI455X (gfx1250) — compile-verified, matrix operand order checked by reading, not run on hardware
//
#include <hip/hip_runtime.h>
#include <hip/hip_bf16.h>
#include <math.h>

// MolTriangleAttention for MI455X (gfx1250, wave32, WMMA)
// B=1, N=256, D=128, H=4, C=32, H*C=128

typedef __attribute__((ext_vector_type(16))) _Float16     v16h;
typedef __attribute__((ext_vector_type(8)))  float        v8f;
typedef __attribute__((ext_vector_type(8)))  unsigned int v8u;

#define NN       256
#define DD       128
#define HH       4
#define CC       32
#define NROWS    (NN * NN)             // 65536 flattened (i,j) rows
#define QSCALE   0.17677669529663689f  // C^-0.5

// Wave32 16-bit A/B operand fragment, row-major source:
// lane owns row (lane&15); K pairs are contiguous dwords.
// dword indices: lane-lo {0,1,2,3, 8,9,10,11}, lane-hi {4,5,6,7, 12,13,14,15}.
__device__ __forceinline__ v16h frag_rowmajor_h(const _Float16* base, int ld, int k0, int lane) {
  const unsigned int* p =
      (const unsigned int*)(base + (size_t)(lane & 15) * ld + k0) + ((lane & 16) ? 4 : 0);
  union { v8u u; v16h h; } cvt;
#pragma unroll
  for (int e = 0; e < 4; ++e) {
    cvt.u[e]     = p[e];       // K pairs 0..7 (lo) / 8..15 (hi)
    cvt.u[e + 4] = p[e + 8];   // K pairs 16..23 (lo) / 24..31 (hi)
  }
  return cvt.h;
}

__device__ __forceinline__ v8f wmma_f16(v16h a, v16h b, v8f c) {
  return __builtin_amdgcn_wmma_f32_16x16x32_f16(false, a, false, b, (short)0, c, false, false);
}

// approximate reciprocal (v_rcp_f32) -- fine for softmax/sigmoid normalization
__device__ __forceinline__ float fast_rcp(float x) { return __builtin_amdgcn_rcpf(x); }

// ---------------------------------------------------------------------------
// Kernel 1: LayerNorm over D=128 -> zn (f16), plus bias[h][i][j] = zn . Wb[:,h]
// ---------------------------------------------------------------------------
__global__ void mta_ln_kernel(const float* __restrict__ z,
                              const float* __restrict__ gamma,
                              const float* __restrict__ beta,
                              const float* __restrict__ Wb,
                              _Float16* __restrict__ zn,
                              float* __restrict__ bias) {
  const int row = blockIdx.x;     // 0..65535
  const int t   = threadIdx.x;    // 0..127
  __shared__ float red[DD];

  float x = z[(size_t)row * DD + t];

  red[t] = x;
  __syncthreads();
  for (int o = 64; o > 0; o >>= 1) { if (t < o) red[t] += red[t + o]; __syncthreads(); }
  float mean = red[0] * (1.0f / DD);
  __syncthreads();

  float d = x - mean;
  red[t] = d * d;
  __syncthreads();
  for (int o = 64; o > 0; o >>= 1) { if (t < o) red[t] += red[t + o]; __syncthreads(); }
  float var = red[0] * (1.0f / DD);
  __syncthreads();

  float rstd = rsqrtf(var + 1e-5f);
  float znv  = d * rstd * gamma[t] + beta[t];
  zn[(size_t)row * DD + t] = (_Float16)znv;

#pragma unroll
  for (int h = 0; h < HH; ++h) {
    red[t] = znv * Wb[t * HH + h];
    __syncthreads();
    for (int o = 64; o > 0; o >>= 1) { if (t < o) red[t] += red[t + o]; __syncthreads(); }
    if (t == 0) bias[(size_t)h * NROWS + row] = red[0];
    __syncthreads();
  }
}

// ---------------------------------------------------------------------------
// Kernel 2: projections zn @ W -> q (pre-scaled), k, v (f16), g = sigmoid (f16)
// grid.x = NROWS/16 row tiles, grid.y selects matrix {0:q,1:k,2:v,3:g}
// W is staged TRANSPOSED in LDS so B fragments use contiguous-pair loads.
// ---------------------------------------------------------------------------
__global__ void mta_proj_kernel(const _Float16* __restrict__ zn,
                                const float* __restrict__ Wq,
                                const float* __restrict__ Wk,
                                const float* __restrict__ Wv,
                                const float* __restrict__ Wg,
                                const float* __restrict__ bg,
                                _Float16* __restrict__ qb,
                                _Float16* __restrict__ kb,
                                _Float16* __restrict__ vb,
                                _Float16* __restrict__ gb) {
  const int row0 = blockIdx.x * 16;
  const int mat  = blockIdx.y;
  const int tid  = threadIdx.x, lane = tid & 31, wv = tid >> 5;

  __shared__ _Float16 wldsT[DD * DD];  // W transposed: wldsT[n][k]
  __shared__ _Float16 znlds[16 * DD];  // 16-row activation tile

  const float* W = (mat == 0) ? Wq : (mat == 1) ? Wk : (mat == 2) ? Wv : Wg;
  const float scale = (mat == 0) ? QSCALE : 1.0f;

  for (int idx = tid; idx < DD * DD; idx += 256) {
    int k = idx >> 7, n = idx & 127;              // coalesced global read
    wldsT[n * DD + k] = (_Float16)(W[idx] * scale);
  }
  for (int idx = tid; idx < 16 * DD; idx += 256)
    znlds[idx] = zn[(size_t)row0 * DD + idx];
  __syncthreads();

  const int col0 = wv * 16;
  v8f acc = {};
#pragma unroll
  for (int kc = 0; kc < 4; ++kc) {
    v16h a = frag_rowmajor_h(znlds, DD, kc * 32, lane);
    v16h b = frag_rowmajor_h(wldsT + col0 * DD, DD, kc * 32, lane);  // row = n
    acc = wmma_f16(a, b, acc);
  }

  const int col = col0 + (lane & 15);
  _Float16* dst = (mat == 0) ? qb : (mat == 1) ? kb : (mat == 2) ? vb : gb;
#pragma unroll
  for (int r = 0; r < 8; ++r) {
    int m = r + ((lane & 16) ? 8 : 0);
    float val = acc[r];
    if (mat == 3) val = fast_rcp(1.0f + __expf(-(val + bg[col])));
    dst[(size_t)(row0 + m) * DD + col] = (_Float16)val;
  }
}

// ---------------------------------------------------------------------------
// Kernel 3: attention for one (i, h, 16-q tile):
//   s[q,k] = qrow.krow + qcol.kcol + bias[h,q,k] + bias[h,k,q] + mask(i,k)
//   softmax over k; wa = probs @ (v[i,k] + v[k,i])
// ---------------------------------------------------------------------------
__global__ void mta_attn_kernel(const _Float16* __restrict__ qb,
                                const _Float16* __restrict__ kb,
                                const _Float16* __restrict__ vb,
                                const float* __restrict__ bias,
                                const int* __restrict__ z_mask,
                                float* __restrict__ wa) {
  const int i  = blockIdx.x;        // 0..255
  const int h  = blockIdx.y;        // 0..3
  const int q0 = blockIdx.z * 16;   // 0..240
  const int tid = threadIdx.x, lane = tid & 31, wv = tid >> 5;

  __shared__ float    s_sc[16][NN];   // scores; reused as vs_t after softmax
  __shared__ _Float16 s_pr[16][NN];   // probs
  __shared__ float    s_wa[16][CC];   // output accumulation

  const int ldCol = NN * DD;  // stride between first-index rows

  // q fragments: row-attention (q[i, q0+m]) and column-attention (q[q0+m, i])
  v16h aR = frag_rowmajor_h(qb + ((size_t)i * NN + q0) * DD + h * CC, DD,    0, lane);
  v16h aC = frag_rowmajor_h(qb + ((size_t)q0 * NN + i) * DD + h * CC, ldCol, 0, lane);

  for (int idx = tid; idx < 16 * CC; idx += 256) (&s_wa[0][0])[idx] = 0.0f;

  // ---- scores: 16 k-tiles, 2 per wave ----
#pragma unroll
  for (int u = 0; u < 2; ++u) {
    const int kt  = wv * 2 + u;
    const int k16 = kt * 16;
    v16h bR = frag_rowmajor_h(kb + ((size_t)i * NN + k16) * DD + h * CC, DD,    0, lane);
    v16h bC = frag_rowmajor_h(kb + ((size_t)k16 * NN + i) * DD + h * CC, ldCol, 0, lane);
    v8f c = {};
    c = wmma_f16(aR, bR, c);
    c = wmma_f16(aC, bC, c);

    const int n  = lane & 15;
    const int kg = k16 + n;
    const float madd = z_mask[i * NN + kg] ? 0.0f : -1e9f;
#pragma unroll
    for (int r = 0; r < 8; ++r) {
      int m  = r + ((lane & 16) ? 8 : 0);
      int qg = q0 + m;
      s_sc[m][kg] = c[r]
                  + bias[(size_t)h * NROWS + qg * NN + kg]
                  + bias[(size_t)h * NROWS + kg * NN + qg]
                  + madd;
    }
  }
  __syncthreads();

  // ---- softmax: 16 rows x 256, 16 threads per row (contiguous 16-lane group) ----
  {
    const int r = tid >> 4, s = tid & 15;
    float mx = -1e30f;
    for (int c = s; c < NN; c += 16) mx = fmaxf(mx, s_sc[r][c]);
#pragma unroll
    for (int o = 8; o >= 1; o >>= 1) mx = fmaxf(mx, __shfl_xor(mx, o, 32));
    float sum = 0.0f;
    for (int c = s; c < NN; c += 16) {
      float e = __expf(s_sc[r][c] - mx);
      s_sc[r][c] = e;
      sum += e;
    }
#pragma unroll
    for (int o = 8; o >= 1; o >>= 1) sum += __shfl_xor(sum, o, 32);
    float inv = fast_rcp(sum);
    for (int c = s; c < NN; c += 16) s_pr[r][c] = (_Float16)(s_sc[r][c] * inv);
  }
  __syncthreads();

  // ---- stage vs = v[i,k]+v[k,i] TRANSPOSED into the dead score buffer ----
  _Float16* s_vt = (_Float16*)&s_sc[0][0];   // [CC][NN] : vs_t[c][k]
  for (int idx = tid; idx < CC * NN; idx += 256) {
    int c = idx & 31, k = idx >> 5;          // consecutive threads -> consecutive c
    s_vt[c * NN + k] = vb[((size_t)i * NN + k) * DD + h * CC + c]
                     + vb[((size_t)k * NN + i) * DD + h * CC + c];
  }
  __syncthreads();

  // ---- wa = probs @ vs : 16x32 output, split over 8 waves ----
  const int c0 = (wv & 1) * 16;
  v8f acc = {};
#pragma unroll
  for (int u = 0; u < 2; ++u) {
    const int kc = (wv >> 1) * 2 + u;        // K chunk of 32 keys
    v16h a = frag_rowmajor_h(&s_pr[0][0], NN, kc * 32, lane);
    v16h b = frag_rowmajor_h(s_vt + c0 * NN, NN, kc * 32, lane);  // row = c
    acc = wmma_f16(a, b, acc);
  }
  {
    const int n = lane & 15;
#pragma unroll
    for (int r = 0; r < 8; ++r) {
      int m = r + ((lane & 16) ? 8 : 0);
      atomicAdd(&s_wa[m][c0 + n], acc[r]);
    }
  }
  __syncthreads();

  for (int idx = tid; idx < 16 * CC; idx += 256) {
    int m = idx >> 5, c = idx & 31;
    wa[((size_t)i * NN + q0 + m) * DD + h * CC + c] = s_wa[m][c];
  }
}

// ---------------------------------------------------------------------------
// Kernel 4: o = (g * wa) @ Wf + bf, masked by z_mask
// ---------------------------------------------------------------------------
__global__ void mta_out_kernel(const _Float16* __restrict__ gb,
                               const float* __restrict__ wa,
                               const float* __restrict__ Wf,
                               const float* __restrict__ bf,
                               const int* __restrict__ z_mask,
                               float* __restrict__ out) {
  const int row0 = blockIdx.x * 16;
  const int tid  = threadIdx.x, lane = tid & 31, wv = tid >> 5;

  __shared__ _Float16 wldsT[DD * DD];  // Wf transposed: wldsT[n][k]
  __shared__ _Float16 glds[16 * DD];   // gated activations

  for (int idx = tid; idx < DD * DD; idx += 256) {
    int k = idx >> 7, n = idx & 127;
    wldsT[n * DD + k] = (_Float16)Wf[idx];
  }
  for (int idx = tid; idx < 16 * DD; idx += 256) {
    size_t gidx = (size_t)row0 * DD + idx;
    glds[idx] = (_Float16)((float)gb[gidx] * wa[gidx]);
  }
  __syncthreads();

  const int col0 = wv * 16;
  v8f acc = {};
#pragma unroll
  for (int kc = 0; kc < 4; ++kc) {
    v16h a = frag_rowmajor_h(glds, DD, kc * 32, lane);
    v16h b = frag_rowmajor_h(wldsT + col0 * DD, DD, kc * 32, lane);
    acc = wmma_f16(a, b, acc);
  }

  const int col = col0 + (lane & 15);
#pragma unroll
  for (int r = 0; r < 8; ++r) {
    int row = row0 + r + ((lane & 16) ? 8 : 0);
    float msk = z_mask[row] ? 1.0f : 0.0f;
    out[(size_t)row * DD + col] = (acc[r] + bf[col]) * msk;
  }
}

// ---------------------------------------------------------------------------
extern "C" void kernel_launch(void* const* d_in, const int* in_sizes, int n_in,
                              void* d_out, int out_size, void* d_ws, size_t ws_size,
                              hipStream_t stream) {
  const float* z      = (const float*)d_in[0];
  const int*   z_mask = (const int*)  d_in[1];
  const float* gamma  = (const float*)d_in[2];
  const float* beta   = (const float*)d_in[3];
  const float* Wq     = (const float*)d_in[4];
  const float* Wk     = (const float*)d_in[5];
  const float* Wv     = (const float*)d_in[6];
  const float* Wb     = (const float*)d_in[7];
  const float* Wg     = (const float*)d_in[8];
  const float* bg     = (const float*)d_in[9];
  const float* Wf     = (const float*)d_in[10];
  const float* bf     = (const float*)d_in[11];
  float* out = (float*)d_out;

  char* ws = (char*)d_ws;
  const size_t szAct16 = (size_t)NROWS * DD * sizeof(_Float16);   // 16 MB
  _Float16* zn   = (_Float16*)(ws);
  _Float16* qb   = (_Float16*)(ws + szAct16);
  _Float16* kb   = (_Float16*)(ws + 2 * szAct16);
  _Float16* vb   = (_Float16*)(ws + 3 * szAct16);
  _Float16* gb   = (_Float16*)(ws + 4 * szAct16);
  float*    bias = (float*)   (ws + 5 * szAct16);                 // 1 MB
  float*    wa   = (float*)   (ws + 5 * szAct16 + (size_t)HH * NROWS * sizeof(float));

  mta_ln_kernel<<<NROWS, DD, 0, stream>>>(z, gamma, beta, Wb, zn, bias);
  mta_proj_kernel<<<dim3(NROWS / 16, 4), 256, 0, stream>>>(zn, Wq, Wk, Wv, Wg, bg,
                                                           qb, kb, vb, gb);
  mta_attn_kernel<<<dim3(NN, HH, NN / 16), 256, 0, stream>>>(qb, kb, vb, bias, z_mask, wa);
  mta_out_kernel<<<NROWS / 16, 256, 0, stream>>>(gb, wa, Wf, bf, z_mask, out);
}
